// MultiboxLoss_70050916598463
// MI455X (gfx1250) — compile-verified, matrix operand order checked by reading, not run on hardware
//
#include <hip/hip_runtime.h>
#include <hip/hip_bf16.h>
#include <hip/hip_fp16.h>

#define N_CLASSES 21
#define NEG_POS_RATIO 3
#define B_DIM 64
#define A_DIM 24564
#define TPB 256

typedef _Float16 v16h __attribute__((ext_vector_type(16)));
typedef float    v8f  __attribute__((ext_vector_type(8)));

// ---------------------------------------------------------------------------
// Wave(32) sum-reduction through the matrix pipe.
// v = hi + lo (two f16s; split error <= 2^-22 * |v|). A(16x32 f16) carries the
// pair at K={0,1} (lanes 0-15) / K={8,9} (lanes 16-31); B = ones(32x16).
// D[m][n] = rowsum(A[m]) = v[m] + v[m+16], accumulated in f32. Summing the 8
// D VGPRs per lane gives half-wave sums; one shfl_xor(16) completes the sum.
// Must be called by ALL lanes (WMMA requires EXEC == all ones).
// ---------------------------------------------------------------------------
__device__ __forceinline__ float wave_reduce_sum_wmma(float v) {
  _Float16 hi = (_Float16)v;
  _Float16 lo = (_Float16)(v - (float)hi);
  v16h a;
#pragma unroll
  for (int i = 0; i < 16; ++i) a[i] = (_Float16)0.0f;
  a[0] = hi;
  a[1] = lo;
  v16h b;
#pragma unroll
  for (int i = 0; i < 16; ++i) b[i] = (_Float16)1.0f;
  v8f c = {};
  c = __builtin_amdgcn_wmma_f32_16x16x32_f16(false, a, false, b, (short)0, c,
                                             false, false);
  float s = c[0] + c[1] + c[2] + c[3] + c[4] + c[5] + c[6] + c[7];
  s += __shfl_xor(s, 16, 32);
  return s; // full wave sum in every lane
}

__device__ __forceinline__ int wave_reduce_sum_i32(int v) {
#pragma unroll
  for (int o = 16; o > 0; o >>= 1) v += __shfl_xor(v, o, 32);
  return v;
}

// Monotone map: float total order -> uint32 total order (ascending).
__device__ __forceinline__ unsigned order_key(float f) {
  unsigned b = __float_as_uint(f);
  return (b & 0x80000000u) ? ~b : (b | 0x80000000u);
}

// ---------------------------------------------------------------------------
// K0: zero accumulators
// ---------------------------------------------------------------------------
__global__ void k_init(float* accs, int* num_pos) {
  int i = threadIdx.x;
  if (i < 2) accs[i] = 0.0f;
  if (i < B_DIM) num_pos[i] = 0;
}

// ---------------------------------------------------------------------------
// K1: one streaming pass over conf/loc.
//   conf tile is staged global->LDS with the CDNA5 async-LDS path
//   (GLOBAL_LOAD_ASYNC_TO_LDS_B128, ASYNCcnt), bypassing VGPRs.
//   per anchor: row-stable LSE over 21 classes, CE = lse - x[target],
//   mining key (positives forced to 0.0), smooth-L1 * pos.
// grid = (ceil(A/256), B), block = 256
// ---------------------------------------------------------------------------
__global__ __launch_bounds__(TPB) void k_per_anchor(
    const float* __restrict__ loc, const float* __restrict__ conf,
    const float* __restrict__ loct, const int* __restrict__ ctgt,
    unsigned* __restrict__ keys, float* __restrict__ perce,
    float* __restrict__ loss_loc_acc, int* __restrict__ num_pos) {
  __shared__ __align__(16) float tile[TPB * N_CLASSES];
  __shared__ float redf[TPB / 32];
  __shared__ int   redi[TPB / 32];

  const int row  = blockIdx.y;
  const int base = blockIdx.x * TPB;
  const int n    = (A_DIM - base < TPB) ? (A_DIM - base) : TPB;
  const long rowoff = (long)row * A_DIM;

  const float* src = conf + (rowoff + base) * N_CLASSES;
  if (blockIdx.x + 1 < gridDim.x) {
    const float* nxt = src + (long)TPB * N_CLASSES;
    __builtin_prefetch(nxt + threadIdx.x * 16, 0, 0);
  }

  // ---- async stage: global -> LDS, 16B per lane per issue ----
  // (n*21 floats is always a multiple of 4 here; both bases are 16B-aligned)
  {
    const int total_bytes = n * N_CLASSES * 4;
    const unsigned lds_base = (unsigned)(uintptr_t)&tile[0];
    const unsigned long long gbase = (unsigned long long)(uintptr_t)src;
    for (int off = threadIdx.x * 16; off < total_bytes; off += TPB * 16) {
      const unsigned           l = lds_base + (unsigned)off;
      const unsigned long long g = gbase + (unsigned long long)off;
      asm volatile("global_load_async_to_lds_b128 %0, %1, off"
                   :: "v"(l), "v"(g) : "memory");
    }
    asm volatile("s_wait_asynccnt 0" ::: "memory");
  }
  __syncthreads();

  float lpart = 0.0f;
  int   posc  = 0;
  if (threadIdx.x < n) {
    const long a   = rowoff + base + threadIdx.x;
    const float* r = &tile[threadIdx.x * N_CLASSES]; // stride 21: conflict-free
    float m = r[0];
#pragma unroll
    for (int i = 1; i < N_CLASSES; ++i) m = fmaxf(m, r[i]);
    float s = 0.0f;
#pragma unroll
    for (int i = 0; i < N_CLASSES; ++i) s += __expf(r[i] - m);
    const float lse = __logf(s) + m;

    const int  tc  = ctgt[a];
    const bool pos = tc > 0;
    const float ce = lse - r[tc];
    perce[a] = ce;
    keys[a]  = order_key(pos ? 0.0f : ce);

    // smooth L1 (beta = 1) on the 4 box coords, masked by pos
    const float4 l4 = ((const float4*)loc)[a];
    const float4 t4 = ((const float4*)loct)[a];
    float sl = 0.0f, d;
    d = fabsf(l4.x - t4.x); sl += (d < 1.0f) ? 0.5f * d * d : d - 0.5f;
    d = fabsf(l4.y - t4.y); sl += (d < 1.0f) ? 0.5f * d * d : d - 0.5f;
    d = fabsf(l4.z - t4.z); sl += (d < 1.0f) ? 0.5f * d * d : d - 0.5f;
    d = fabsf(l4.w - t4.w); sl += (d < 1.0f) ? 0.5f * d * d : d - 0.5f;
    if (pos) { lpart = sl; posc = 1; }
  }

  // block reductions (convergent: every thread participates)
  const float ws = wave_reduce_sum_wmma(lpart);
  const int   wi = wave_reduce_sum_i32(posc);
  const int lane = threadIdx.x & 31, wid = threadIdx.x >> 5;
  if (lane == 0) { redf[wid] = ws; redi[wid] = wi; }
  __syncthreads();
  if (threadIdx.x == 0) {
    float tf = 0.0f; int ti = 0;
#pragma unroll
    for (int i = 0; i < TPB / 32; ++i) { tf += redf[i]; ti += redi[i]; }
    atomicAdd(loss_loc_acc, tf);
    atomicAdd(&num_pos[row], ti);
  }
}

// ---------------------------------------------------------------------------
// K2: per-row hard-negative mining + masked CE sum.
//   radix-select (MSB-first, 8 bits/pass, uint4 reads) the num_neg-th largest
//   key, then a stable index-ordered quota pass for ties => exact
//   argsort-rank semantics. keys/perce rows (98 KB each) are L2-resident.
// grid = B, block = 256
// ---------------------------------------------------------------------------
__global__ __launch_bounds__(TPB) void k_mine(
    const unsigned* __restrict__ keys, const float* __restrict__ perce,
    const int* __restrict__ ctgt, const int* __restrict__ num_pos,
    float* __restrict__ loss_conf_acc) {
  __shared__ unsigned hist[256];
  __shared__ int      sh_sel[2];
  __shared__ int      cnts[TPB];
  __shared__ float    redf[TPB / 32];

  const int  row    = blockIdx.x;
  const long rowoff = (long)row * A_DIM;
  const unsigned* krow = keys + rowoff;
  const int  np     = num_pos[row];
  int num_neg = NEG_POS_RATIO * np;
  if (num_neg > A_DIM - 1) num_neg = A_DIM - 1;
  const bool have = (num_neg > 0);

  unsigned prefix = 0, pmask = 0;
  int k = num_neg;
  if (have) {
    for (int shift = 24; shift >= 0; shift -= 8) {
      for (int i = threadIdx.x; i < 256; i += TPB) hist[i] = 0u;
      __syncthreads();
      // A_DIM % 4 == 0 and the row base is 16B-aligned: uint4 reads
      for (int i = threadIdx.x; i < A_DIM / 4; i += TPB) {
        const uint4 u4 = ((const uint4*)krow)[i];
        if ((u4.x & pmask) == prefix) atomicAdd(&hist[(u4.x >> shift) & 255u], 1u);
        if ((u4.y & pmask) == prefix) atomicAdd(&hist[(u4.y >> shift) & 255u], 1u);
        if ((u4.z & pmask) == prefix) atomicAdd(&hist[(u4.z >> shift) & 255u], 1u);
        if ((u4.w & pmask) == prefix) atomicAdd(&hist[(u4.w >> shift) & 255u], 1u);
      }
      __syncthreads();
      if (threadIdx.x == 0) {
        int c = 0, chosen = 0, krem = k;
        for (int bb = 255; bb >= 0; --bb) {
          int h = (int)hist[bb];
          if (c + h >= k) { chosen = bb; krem = k - c; break; }
          c += h;
        }
        sh_sel[0] = chosen;
        sh_sel[1] = krem;
      }
      __syncthreads();
      prefix |= ((unsigned)sh_sel[0]) << shift;
      pmask  |= 255u << shift;
      k = sh_sel[1];
      __syncthreads();
    }
  }
  const unsigned tkey  = prefix; // key of the num_neg-th largest element
  const int      quota = k;      // how many ties (by ascending index) to keep

  // stable tie handling: contiguous chunk per thread; exclusive scan of
  // per-chunk tie counts gives each thread its starting tie-rank.
  const int L     = (A_DIM + TPB - 1) / TPB; // 96
  const int begin = threadIdx.x * L;
  const int end   = (begin + L < A_DIM) ? (begin + L) : A_DIM;
  int myeq = 0;
  if (have)
    for (int i = begin; i < end; ++i) myeq += (krow[i] == tkey) ? 1 : 0;
  cnts[threadIdx.x] = myeq;
  __syncthreads();
  if (threadIdx.x == 0) {
    int run = 0;
    for (int i = 0; i < TPB; ++i) { int c = cnts[i]; cnts[i] = run; run += c; }
  }
  __syncthreads();

  int running = cnts[threadIdx.x];
  float sum_local = 0.0f;
  for (int i = begin; i < end; ++i) {
    const long idx = rowoff + i;
    bool sel = ctgt[idx] > 0; // positives always included (mask = pos | neg)
    if (have) {
      unsigned u = krow[i];
      if (u > tkey) sel = true;
      else if (u == tkey) { if (running < quota) sel = true; ++running; }
    }
    if (sel) sum_local += perce[idx];
  }

  const float ws = wave_reduce_sum_wmma(sum_local);
  const int lane = threadIdx.x & 31, wid = threadIdx.x >> 5;
  if (lane == 0) redf[wid] = ws;
  __syncthreads();
  if (threadIdx.x == 0) {
    float t = 0.0f;
#pragma unroll
    for (int i = 0; i < TPB / 32; ++i) t += redf[i];
    atomicAdd(loss_conf_acc, t);
  }
}

// ---------------------------------------------------------------------------
// K3: finalize
// ---------------------------------------------------------------------------
__global__ void k_final(const float* __restrict__ accs,
                        const int* __restrict__ num_pos,
                        float* __restrict__ out) {
  if (threadIdx.x == 0) {
    int tp = 0;
    for (int i = 0; i < B_DIM; ++i) tp += num_pos[i];
    const float n  = (float)tp;
    const float lc = accs[1] / n; // loss_conf
    const float ll = accs[0] / n; // loss_loc
    out[0] = lc;
    out[1] = ll;
    out[2] = ll + lc;
  }
}

// ---------------------------------------------------------------------------
// Host launcher. Workspace layout:
//   [0,   64)  : float accs[2]      (loc_acc, conf_acc)
//   [64, 1024) : int   num_pos[64]
//   [1024, +K) : uint  keys[B*A]
//   [.., +K)   : float perce[B*A]
// total ~12.6 MB
// ---------------------------------------------------------------------------
extern "C" void kernel_launch(void* const* d_in, const int* in_sizes, int n_in,
                              void* d_out, int out_size, void* d_ws,
                              size_t ws_size, hipStream_t stream) {
  const float* loc  = (const float*)d_in[0];
  const float* conf = (const float*)d_in[1];
  const float* loct = (const float*)d_in[2];
  const int*   ctgt = (const int*)d_in[3];
  float* out = (float*)d_out;

  char* ws = (char*)d_ws;
  float*    accs    = (float*)ws;
  int*      num_pos = (int*)(ws + 64);
  unsigned* keys    = (unsigned*)(ws + 1024);
  float*    perce   = (float*)(ws + 1024 + sizeof(unsigned) * (size_t)B_DIM * A_DIM);

  k_init<<<1, 256, 0, stream>>>(accs, num_pos);

  dim3 grid((A_DIM + TPB - 1) / TPB, B_DIM);
  k_per_anchor<<<grid, TPB, 0, stream>>>(loc, conf, loct, ctgt, keys, perce,
                                         &accs[0], num_pos);

  k_mine<<<B_DIM, TPB, 0, stream>>>(keys, perce, ctgt, num_pos, &accs[1]);

  k_final<<<1, 64, 0, stream>>>(accs, num_pos, out);
}
